// VectorQuantizer_128849019168
// MI455X (gfx1250) — matrix operand order repaired and checked by reading, not compiled, not run on hardware
//
#include <hip/hip_runtime.h>
#include <hip/hip_bf16.h>

typedef __attribute__((ext_vector_type(16))) __bf16       v16bf;
typedef __attribute__((ext_vector_type(8)))  float        v8f;
typedef __attribute__((ext_vector_type(8)))  unsigned int v8u;
typedef unsigned int   u32;
typedef unsigned short u16;

// ---- helpers -------------------------------------------------------------
__device__ __forceinline__ u16 f2bf(float f) {
    u32 u = __float_as_uint(f);
    u32 r = u + 0x7FFFu + ((u >> 16) & 1u);   // round-to-nearest-even
    return (u16)(r >> 16);
}
__device__ __forceinline__ u32 pack2(float lo, float hi) {
    return (u32)f2bf(lo) | ((u32)f2bf(hi) << 16);
}

// ---- kernel 0: zero histogram + loss accumulator -------------------------
__global__ void zero_kernel(u32* counts, float* accum) {
    int t = threadIdx.x;
    for (int k = t; k < 1024; k += 256) counts[k] = 0u;
    if (t == 0) accum[0] = 0.0f;
}

// ---- kernel 1: codebook -> bf16, plus ||e||^2 ----------------------------
__global__ void prep_kernel(const float* __restrict__ emb,
                            u16* __restrict__ emb_bf,
                            float* __restrict__ e2) {
    int k = blockIdx.x;            // 1024 blocks, 32 threads (one wave)
    int t = threadIdx.x;
    int c0 = t * 8;
    const float4* src = (const float4*)(emb + k * 256 + c0);
    float4 a = src[0], b = src[1];
    uint4 pk;
    pk.x = pack2(a.x, a.y); pk.y = pack2(a.z, a.w);
    pk.z = pack2(b.x, b.y); pk.w = pack2(b.z, b.w);
    *(uint4*)(emb_bf + k * 256 + c0) = pk;
    float ss = a.x*a.x + a.y*a.y + a.z*a.z + a.w*a.w
             + b.x*b.x + b.y*b.y + b.z*b.z + b.w*b.w;
    #pragma unroll
    for (int off = 16; off >= 1; off >>= 1) ss += __shfl_xor(ss, off, 32);
    if (t == 0) e2[k] = ss;
}

// ---- kernel 2: WMMA distance GEMM + fused argmax -------------------------
// grid 256 blocks x 256 threads; block owns 128 rows, wave owns 16 rows.
// Codebook chunks (128 codes, 66KB) are double-buffered in LDS and filled
// with GLOBAL_LOAD_ASYNC_TO_LDS_B128 overlapped with the WMMA compute.
__global__ void __launch_bounds__(256)
vq_kernel(const float* __restrict__ z, const u16* __restrict__ emb_bf,
          const float* __restrict__ e2, int* __restrict__ idx_out) {
    __shared__ u16 lds_emb[2][128 * 264];  // rows padded to 528B (bank-friendly)
    const int t    = threadIdx.x;
    const int wave = t >> 5, lane = t & 31;
    const int hf   = lane >> 4, mcol = lane & 15;

    // A fragments: this lane holds row M = lane%16 of its wave's 16-row tile.
    // z layout (B,C,H,W): row n, channel k lives at base(n) + k*1024.
    const int n     = blockIdx.x * 128 + wave * 16 + mcol;
    const int abase = (n >> 10) * 262144 + (n & 1023);
    v16bf A[8];
    #pragma unroll
    for (int c = 0; c < 8; ++c) {
        const int k0 = c * 32 + 8 * hf;          // 16-bit A 16x32 VGPR layout
        float f[16];
        #pragma unroll
        for (int i = 0; i < 16; ++i) {
            int koff = k0 + ((i < 8) ? i : i + 8);   // K = k0+{0..7}, k0+16+{0..7}
            f[i] = z[abase + koff * 1024];
        }
        v8u pw;
        #pragma unroll
        for (int j = 0; j < 8; ++j) pw[j] = pack2(f[2 * j], f[2 * j + 1]);
        A[c] = __builtin_bit_cast(v16bf, pw);
    }

    // async stage of one 128-code chunk into LDS buffer `buf`
    const unsigned long long gbase0 = (unsigned long long)(size_t)emb_bf;
    auto stage = [&](int chunk, int buf) {
        unsigned long long gb = gbase0 + (unsigned long long)chunk * 65536ull;
        #pragma unroll
        for (int i = 0; i < 16; ++i) {
            int idx16 = i * 256 + t;                 // 16B unit within chunk
            int cr = idx16 >> 5, col = idx16 & 31;
            unsigned lds_b = (unsigned)(size_t)&lds_emb[buf][cr * 264 + col * 8];
            unsigned voff  = (unsigned)(idx16 * 16);
            asm volatile("global_load_async_to_lds_b128 %0, %1, %2"
                         :: "v"(lds_b), "v"(voff), "s"(gb) : "memory");
        }
    };

    stage(0, 0);
    asm volatile("s_wait_asynccnt 0x0" ::: "memory");
    __syncthreads();

    float bestv[8]; int besti[8];
    #pragma unroll
    for (int r = 0; r < 8; ++r) { bestv[r] = -3.0e38f; besti[r] = 0; }

    for (int chunk = 0; chunk < 8; ++chunk) {     // 8 x 128 codes
        const int buf = chunk & 1;
        if (chunk < 7) stage(chunk + 1, buf ^ 1); // overlap next stage w/ WMMA
        float e2c[8];                              // per-chunk bias prefetch
        #pragma unroll
        for (int tile = 0; tile < 8; ++tile)
            e2c[tile] = e2[chunk * 128 + tile * 16 + mcol];

        for (int tile = 0; tile < 8; ++tile) {    // 16 codes per tile
            const int codeL = tile * 16 + mcol;
            v8f acc = {0.f, 0.f, 0.f, 0.f, 0.f, 0.f, 0.f, 0.f};
            #pragma unroll
            for (int c = 0; c < 8; ++c) {
                // 16-bit B 32x16: lane holds col N=lane%16, K=8*hf+{0..7}, 16+8*hf+{0..7}
                const u16* bp = &lds_emb[buf][codeL * 264 + c * 32 + 8 * hf];
                uint4 b0 = *(const uint4*)(bp);
                uint4 b1 = *(const uint4*)(bp + 16);
                v8u bw;
                bw[0] = b0.x; bw[1] = b0.y; bw[2] = b0.z; bw[3] = b0.w;
                bw[4] = b1.x; bw[5] = b1.y; bw[6] = b1.z; bw[7] = b1.w;
                acc = __builtin_amdgcn_wmma_f32_16x16x32_bf16(
                        false, A[c], false, __builtin_bit_cast(v16bf, bw),
                        (short)0, acc, false, false);
            }
            const float bias = 0.5f * e2c[tile];   // argmax(z.e - 0.5||e||^2)
            const int codeG = chunk * 128 + codeL;
            #pragma unroll
            for (int r = 0; r < 8; ++r) {          // C layout: VGPR r = row r+8*hf
                float s = acc[r] - bias;
                if (s > bestv[r]) { bestv[r] = s; besti[r] = codeG; }
            }
        }
        asm volatile("s_wait_asynccnt 0x0" ::: "memory");
        __syncthreads();   // staged buffer ready; all waves done with current
    }

    // reduce across the 16 lanes that hold the same row (different codes)
    #pragma unroll
    for (int m = 8; m >= 1; m >>= 1) {
        #pragma unroll
        for (int r = 0; r < 8; ++r) {
            float ov = __shfl_xor(bestv[r], m, 32);
            int   oi = __shfl_xor(besti[r], m, 32);
            if (ov > bestv[r] || (ov == bestv[r] && oi < besti[r])) {
                bestv[r] = ov; besti[r] = oi;
            }
        }
    }
    if (mcol == 0) {
        #pragma unroll
        for (int r = 0; r < 8; ++r)
            idx_out[blockIdx.x * 128 + wave * 16 + hf * 8 + r] = besti[r];
    }
}

// ---- kernel 3: gather codes, write transposed output, loss + histogram ---
__global__ void __launch_bounds__(256)
scatter_kernel(const float* __restrict__ z, const float* __restrict__ emb,
               const int* __restrict__ idx, u32* __restrict__ counts,
               float* __restrict__ accum, float* __restrict__ out) {
    __shared__ int   sidx[16];
    __shared__ float semb[16][260];
    __shared__ float red[8];
    const int t  = threadIdx.x;
    const int n0 = blockIdx.x * 16;                       // 16 rows per block
    const int base = (n0 >> 10) * 262144 + (n0 & 1023);   // (b, h, w0) offset
    if (t < 16) {
        int id = idx[n0 + t];
        sidx[t] = id;
        atomicAdd(&counts[id], 1u);
    }
    __syncthreads();
    {   // gather 16 fp32 codebook rows into LDS (exact values for out/loss)
        int row = t >> 4, c0 = (t & 15) * 16;
        const float4* src = (const float4*)(emb + sidx[row] * 256 + c0);
        float4* dst = (float4*)&semb[row][c0];
        dst[0] = src[0]; dst[1] = src[1]; dst[2] = src[2]; dst[3] = src[3];
    }
    __syncthreads();
    float local = 0.0f;
    #pragma unroll 4
    for (int pass = 0; pass < 16; ++pass) {
        int c = pass * 16 + (t >> 4);
        int m = t & 15;
        int addr = base + c * 1024 + m;       // coalesced along w
        float e  = semb[m][c];
        float zv = z[addr];
        out[addr] = e;                        // straight-through forward value
        float d = e - zv;
        local += d * d;
    }
    #pragma unroll
    for (int off = 16; off >= 1; off >>= 1) local += __shfl_xor(local, off, 32);
    if ((t & 31) == 0) red[t >> 5] = local;
    __syncthreads();
    if (t == 0) {
        float s = 0.f;
        for (int w = 0; w < 8; ++w) s += red[w];
        atomicAdd(accum, s);
    }
}

// ---- kernel 4: scalars ---------------------------------------------------
__global__ void finalize_kernel(const float* __restrict__ accum,
                                const u32* __restrict__ counts,
                                float* __restrict__ scal) {
    __shared__ float red[8];
    const int t = threadIdx.x;   // 256 threads, 4 codes each
    float s = 0.0f;
    #pragma unroll
    for (int j = 0; j < 4; ++j) {
        float p = (float)counts[t * 4 + j] * (1.0f / 32768.0f);
        s += p * logf(p + 1e-10f);
    }
    #pragma unroll
    for (int off = 16; off >= 1; off >>= 1) s += __shfl_xor(s, off, 32);
    if ((t & 31) == 0) red[t >> 5] = s;
    __syncthreads();
    if (t == 0) {
        float tot = 0.f;
        for (int w = 0; w < 8; ++w) tot += red[w];
        scal[0] = 1.25f * accum[0] * (1.0f / 8388608.0f);  // q + 0.25*e latent
        scal[1] = expf(-tot);                               // perplexity
    }
}

// ---- host launcher -------------------------------------------------------
extern "C" void kernel_launch(void* const* d_in, const int* in_sizes, int n_in,
                              void* d_out, int out_size, void* d_ws, size_t ws_size,
                              hipStream_t stream) {
    const float* z   = (const float*)d_in[0];   // (32,256,32,32) fp32
    const float* emb = (const float*)d_in[1];   // (1024,256) fp32
    float* out = (float*)d_out;                 // 8388608 image + 2 scalars

    char* ws      = (char*)d_ws;
    u16*   emb_bf = (u16*)(ws);                 // 524288 B
    float* e2     = (float*)(ws + 524288);      //   4096 B
    int*   idxb   = (int*)(ws + 528384);        // 131072 B
    u32*   counts = (u32*)(ws + 659456);        //   4096 B
    float* accum  = (float*)(ws + 663552);      //      4 B

    zero_kernel    <<<   1, 256, 0, stream>>>(counts, accum);
    prep_kernel    <<<1024,  32, 0, stream>>>(emb, emb_bf, e2);
    vq_kernel      <<< 256, 256, 0, stream>>>(z, emb_bf, e2, idxb);
    scatter_kernel <<<2048, 256, 0, stream>>>(z, emb, idxb, counts, accum, out);
    finalize_kernel<<<   1, 256, 0, stream>>>(accum, counts, out + 8388608);
}
